// MambaResBlock_28467043238500
// MI455X (gfx1250) — matrix operand order repaired and checked by reading, not compiled, not run on hardware
//
#include <hip/hip_runtime.h>
#include <hip/hip_bf16.h>

typedef __attribute__((ext_vector_type(16))) _Float16 v16h;
typedef __attribute__((ext_vector_type(8)))  _Float16 v8h;
typedef __attribute__((ext_vector_type(4)))  _Float16 v4h;
typedef __attribute__((ext_vector_type(8)))  float    v8f;

#define LN_EPS 1e-3f

static __device__ __forceinline__ float wave_reduce_sum(float v) {
#pragma unroll
    for (int off = 16; off > 0; off >>= 1)
        v += __shfl_xor(v, off, 32);
    return v;
}

// ---------------------------------------------------------------------------
// Pack a K x N row-major f32 weight matrix into per-lane WMMA B-fragment
// layout (f16): index ((nt*(K/32)+kc)*32 + lane)*16 + i  holds
// W[kc*32 + (lane>>4)*8 + (i&7) + (i>>3)*16][nt*16 + (lane&15)].
// ---------------------------------------------------------------------------
__global__ void pack_b_kernel(const float* __restrict__ W, _Float16* __restrict__ P,
                              int K, int N) {
    int t = blockIdx.x * blockDim.x + threadIdx.x;
    if (t >= K * N) return;
    int i    = t & 15;
    int lane = (t >> 4) & 31;
    int rest = t >> 9;                 // / (16*32)
    int kchunks = K >> 5;
    int kc = rest % kchunks;
    int nt = rest / kchunks;
    int k = kc * 32 + ((lane >> 4) << 3) + (i & 7) + ((i >> 3) << 4);
    int n = (nt << 4) + (lane & 15);
    P[t] = (_Float16)W[(size_t)k * N + n];
}

// ---------------------------------------------------------------------------
// LayerNorm over din=128, one wave per token, output f16 for WMMA A operand.
// ---------------------------------------------------------------------------
__global__ __launch_bounds__(256) void ln_kernel(const float* __restrict__ x,
                                                 const float* __restrict__ gamma,
                                                 const float* __restrict__ beta,
                                                 _Float16* __restrict__ h16) {
    int token = blockIdx.x * 8 + (threadIdx.x >> 5);
    int lane  = threadIdx.x & 31;
    const float4 v = ((const float4*)(x + (size_t)token * 128))[lane];
    float vv[4] = {v.x, v.y, v.z, v.w};
    float s  = vv[0] + vv[1] + vv[2] + vv[3];
    float sq = vv[0]*vv[0] + vv[1]*vv[1] + vv[2]*vv[2] + vv[3]*vv[3];
    s  = wave_reduce_sum(s);
    sq = wave_reduce_sum(sq);
    float mean = s * (1.0f / 128.0f);
    float var  = sq * (1.0f / 128.0f) - mean * mean;
    float inv  = rsqrtf(var + LN_EPS);
    v4h o;
#pragma unroll
    for (int i = 0; i < 4; ++i) {
        int c = lane * 4 + i;
        o[i] = (_Float16)((vv[i] - mean) * inv * gamma[c] + beta[c]);
    }
    *(v4h*)(h16 + (size_t)token * 128 + lane * 4) = o;
}

// ---------------------------------------------------------------------------
// WMMA GEMM: Out(MxN) = act(A(MxK,f16) @ Wpacked + bias) [+ resid]
// One wave computes a 16 x (16*NT) tile: A fragment loaded once per k-chunk
// and reused across NT v_wmma (4x arithmetic intensity on A).
// ACT: 0=none, 1=SiLU, 2=softplus.  RES: add residual (f32, MxN).
// ---------------------------------------------------------------------------
template <int KDIM, int NT, int ACT, bool RES>
__global__ __launch_bounds__(256) void gemm_wmma(const _Float16* __restrict__ A,
                                                 const _Float16* __restrict__ Bp,
                                                 const float* __restrict__ bias,
                                                 const float* __restrict__ resid,
                                                 float* __restrict__ Out,
                                                 int N) {
    const int lane = threadIdx.x & 31;
    const int wave = threadIdx.x >> 5;
    const int ngroups = N / (16 * NT);
    const int tile = blockIdx.x * 8 + wave;
    const int mt  = tile / ngroups;
    const int nt0 = (tile % ngroups) * NT;
    const int row = lane & 15;
    const int hi  = lane >> 4;

    v8f acc[NT];
#pragma unroll
    for (int j = 0; j < NT; ++j) acc[j] = (v8f){};

    const _Float16* arow = A + ((size_t)(mt * 16 + row)) * KDIM + hi * 8;
#pragma unroll
    for (int kc = 0; kc < KDIM / 32; ++kc) {
        // A fragment 16x32 f16 (lanes 0-15: K 0-7,16-23; lanes 16-31: K 8-15,24-31)
        v8h alo = *(const v8h*)(arow + kc * 32);
        v8h ahi = *(const v8h*)(arow + kc * 32 + 16);
        v16h a;
#pragma unroll
        for (int i = 0; i < 8; ++i) { a[i] = alo[i]; a[8 + i] = ahi[i]; }
#pragma unroll
        for (int j = 0; j < NT; ++j) {
            const _Float16* bptr =
                Bp + ((size_t)(nt0 + j) * (KDIM / 32) * 32 + (size_t)kc * 32 + lane) * 16;
            v16h b = *(const v16h*)bptr;
            acc[j] = __builtin_amdgcn_wmma_f32_16x16x32_f16(false, a, false, b,
                                                            (short)0, acc[j], false, false);
        }
    }
    const size_t rowbase = (size_t)(mt * 16 + hi * 8) * N + nt0 * 16 + row;
#pragma unroll
    for (int j = 0; j < NT; ++j) {
        const float bv = bias[nt0 * 16 + j * 16 + row];
#pragma unroll
        for (int r = 0; r < 8; ++r) {
            size_t o = rowbase + (size_t)r * N + j * 16;
            float v = acc[j][r] + bv;
            if (ACT == 1) v = v / (1.0f + __expf(-v));                  // SiLU
            else if (ACT == 2) v = (v > 20.0f) ? v : log1pf(__expf(v)); // softplus
            if (RES) v += resid[o];
            Out[o] = v;
        }
    }
}

// ---------------------------------------------------------------------------
// Depthwise causal conv (k=4) + SiLU; emits f32 (scan) and f16 (GEMM) copies.
// ---------------------------------------------------------------------------
__global__ __launch_bounds__(256) void conv_silu_kernel(const float* __restrict__ x1pre,
                                                        const float* __restrict__ cw,
                                                        const float* __restrict__ cb,
                                                        float* __restrict__ x1,
                                                        _Float16* __restrict__ x116,
                                                        int L) {
    int t = blockIdx.x * 256 + threadIdx.x;
    int d = t & 255;
    int l = (t >> 8) % L;
    int bb = t / (L << 8);
    size_t base = ((size_t)bb * L) << 8;
    float acc = cb[d];
#pragma unroll
    for (int w = 0; w < 4; ++w) {
        int ll = l - 3 + w;
        if (ll >= 0) acc += cw[w * 256 + d] * x1pre[base + ((size_t)ll << 8) + d];
    }
    float sv = acc / (1.0f + __expf(-acc));
    size_t o = base + ((size_t)l << 8) + d;
    x1[o] = sv;
    x116[o] = (_Float16)sv;
}

// ---------------------------------------------------------------------------
// Chunked selective scan (s_l = a_l*s_{l-1} + b_l is an associative scan).
// CH = chunk length, NC = L/CH chunks. One wave per (b, d, chunk); lane = state n.
// Pass 1: local scan from s=0, plus decay product P = prod(a_l). No reduction.
// ---------------------------------------------------------------------------
#define SCAN_CH 128
#define SCAN_NC 16   // 2048 / 128

__global__ __launch_bounds__(256) void scan_pass1(const float* __restrict__ x1,
                                                  const float* __restrict__ delta,
                                                  const float* __restrict__ Bm,
                                                  const float* __restrict__ A_log,
                                                  float* __restrict__ Sbuf,
                                                  float* __restrict__ Pbuf,
                                                  int L) {
    const int wid = blockIdx.x * 8 + (threadIdx.x >> 5);
    const int n   = threadIdx.x & 31;
    const int bd  = wid / SCAN_NC;          // (b*256 + d)
    const int c   = wid % SCAN_NC;
    const int d   = bd & 255;
    const int bb  = bd >> 8;
    const float An = -__expf(A_log[d * 32 + n]);
    float s = 0.0f, P = 1.0f;
    const size_t tok0 = (size_t)bb * L + c * SCAN_CH;
    for (int l = 0; l < SCAN_CH; ++l) {
        size_t tok = tok0 + l;
        if (l + 8 < SCAN_CH) {
            __builtin_prefetch(&delta[(tok + 8) * 256 + d], 0, 1);
            __builtin_prefetch(&Bm[(tok + 8) * 32 + n], 0, 1);
        }
        float dv = delta[tok * 256 + d];
        float u  = x1[tok * 256 + d];
        float Bn = Bm[tok * 32 + n];
        float a  = __expf(dv * An);
        s = s * a + dv * u * Bn;
        P *= a;
    }
    size_t o = ((size_t)bd * SCAN_NC + c) * 32 + n;
    Sbuf[o] = s;
    Pbuf[o] = P;
}

// Sequential carry across the NC chunks: 32768 independent (b,d,n) threads.
__global__ __launch_bounds__(256) void scan_combine(const float* __restrict__ Sbuf,
                                                    const float* __restrict__ Pbuf,
                                                    float* __restrict__ Init) {
    int tid = blockIdx.x * 256 + threadIdx.x;   // (b*256+d)*32 + n
    int bd = tid >> 5;
    int n  = tid & 31;
    float carry = 0.0f;
#pragma unroll
    for (int c = 0; c < SCAN_NC; ++c) {
        size_t o = ((size_t)bd * SCAN_NC + c) * 32 + n;
        Init[o] = carry;
        carry = Pbuf[o] * carry + Sbuf[o];
    }
}

// Pass 2: re-run recurrence from correct chunk-start state; emit
// y = sum_n(s*C) + u*D, gated by silu(h@w2+b2), converted to f16 for the w3 GEMM.
__global__ __launch_bounds__(256) void scan_pass2(const float* __restrict__ x1,
                                                  const float* __restrict__ delta,
                                                  const float* __restrict__ Bm,
                                                  const float* __restrict__ Cm,
                                                  const float* __restrict__ A_log,
                                                  const float* __restrict__ Dp,
                                                  const float* __restrict__ g,
                                                  const float* __restrict__ Init,
                                                  _Float16* __restrict__ prod16,
                                                  int L) {
    const int wid = blockIdx.x * 8 + (threadIdx.x >> 5);
    const int n   = threadIdx.x & 31;
    const int bd  = wid / SCAN_NC;
    const int c   = wid % SCAN_NC;
    const int d   = bd & 255;
    const int bb  = bd >> 8;
    const float An = -__expf(A_log[d * 32 + n]);
    const float Dd = Dp[d];
    float s = Init[((size_t)bd * SCAN_NC + c) * 32 + n];
    const size_t tok0 = (size_t)bb * L + c * SCAN_CH;
    for (int l = 0; l < SCAN_CH; ++l) {
        size_t tok = tok0 + l;
        if (l + 8 < SCAN_CH) {
            __builtin_prefetch(&delta[(tok + 8) * 256 + d], 0, 1);
            __builtin_prefetch(&Cm[(tok + 8) * 32 + n], 0, 1);
        }
        float dv = delta[tok * 256 + d];
        float u  = x1[tok * 256 + d];
        float Bn = Bm[tok * 32 + n];
        float Cn = Cm[tok * 32 + n];
        s = s * __expf(dv * An) + dv * u * Bn;
        float y = wave_reduce_sum(s * Cn) + u * Dd;
        if (n == 0) {
            float gv = g[tok * 256 + d];
            prod16[tok * 256 + d] = (_Float16)(y * gv);
        }
    }
}

// ---------------------------------------------------------------------------
extern "C" void kernel_launch(void* const* d_in, const int* in_sizes, int n_in,
                              void* d_out, int out_size, void* d_ws, size_t ws_size,
                              hipStream_t stream) {
    (void)in_sizes; (void)n_in; (void)out_size; (void)ws_size;
    const float* x        = (const float*)d_in[0];
    const float* ln_gamma = (const float*)d_in[1];
    const float* ln_beta  = (const float*)d_in[2];
    const float* w1       = (const float*)d_in[3];
    const float* b1       = (const float*)d_in[4];
    const float* conv_w   = (const float*)d_in[5];
    const float* conv_b   = (const float*)d_in[6];
    const float* w2       = (const float*)d_in[7];
    const float* b2       = (const float*)d_in[8];
    const float* wB       = (const float*)d_in[9];
    const float* bB       = (const float*)d_in[10];
    const float* wC       = (const float*)d_in[11];
    const float* bC       = (const float*)d_in[12];
    const float* wd       = (const float*)d_in[13];
    const float* bd       = (const float*)d_in[14];
    const float* A_log    = (const float*)d_in[15];
    const float* Dv       = (const float*)d_in[16];
    const float* w3       = (const float*)d_in[17];
    const float* b3       = (const float*)d_in[18];

    const int B = 4, L = 2048, DIN = 128, DP = 256, NS = 32;
    const int M = B * L;  // 8192 tokens

    char* ws = (char*)d_ws;
    size_t off = 0;
    auto alloc = [&](size_t bytes) -> void* {
        off = (off + 255) & ~(size_t)255;
        void* p = ws + off;
        off += bytes;
        return p;
    };
    _Float16* h16    = (_Float16*)alloc((size_t)M * DIN * 2);
    _Float16* w1p    = (_Float16*)alloc((size_t)DIN * DP * 2);
    _Float16* w2p    = (_Float16*)alloc((size_t)DIN * DP * 2);
    _Float16* wBp    = (_Float16*)alloc((size_t)DP * NS * 2);
    _Float16* wCp    = (_Float16*)alloc((size_t)DP * NS * 2);
    _Float16* wdp    = (_Float16*)alloc((size_t)DP * DP * 2);
    _Float16* w3p    = (_Float16*)alloc((size_t)DP * DIN * 2);
    float*    x1pre  = (float*)alloc((size_t)M * DP * 4);
    float*    x1     = (float*)alloc((size_t)M * DP * 4);
    _Float16* x116   = (_Float16*)alloc((size_t)M * DP * 2);
    float*    gbuf   = (float*)alloc((size_t)M * DP * 4);   // silu(h@w2+b2)
    float*    Bmat   = (float*)alloc((size_t)M * NS * 4);
    float*    Cmat   = (float*)alloc((size_t)M * NS * 4);
    float*    dbuf   = (float*)alloc((size_t)M * DP * 4);   // softplus(x1@wd+bd)
    _Float16* prod16 = (_Float16*)alloc((size_t)M * DP * 2);
    float*    Sbuf   = (float*)alloc((size_t)B * DP * SCAN_NC * NS * 4);
    float*    Pbuf   = (float*)alloc((size_t)B * DP * SCAN_NC * NS * 4);
    float*    Ibuf   = (float*)alloc((size_t)B * DP * SCAN_NC * NS * 4);

    // Pack weights into WMMA B-fragment layout (f16)
    pack_b_kernel<<<(DIN * DP + 255) / 256, 256, 0, stream>>>(w1, w1p, DIN, DP);
    pack_b_kernel<<<(DIN * DP + 255) / 256, 256, 0, stream>>>(w2, w2p, DIN, DP);
    pack_b_kernel<<<(DP * NS + 255) / 256, 256, 0, stream>>>(wB, wBp, DP, NS);
    pack_b_kernel<<<(DP * NS + 255) / 256, 256, 0, stream>>>(wC, wCp, DP, NS);
    pack_b_kernel<<<(DP * DP + 255) / 256, 256, 0, stream>>>(wd, wdp, DP, DP);
    pack_b_kernel<<<(DP * DIN + 255) / 256, 256, 0, stream>>>(w3, w3p, DP, DIN);

    // LayerNorm -> h (f16)
    ln_kernel<<<M / 8, 256, 0, stream>>>(x, ln_gamma, ln_beta, h16);

    // x1pre = h @ w1 + b1          (16x64 tiles, 4-way A reuse)
    gemm_wmma<128, 4, 0, false><<<(M / 16) * (DP / 64) / 8, 256, 0, stream>>>(
        h16, w1p, b1, nullptr, x1pre, DP);
    // gate = silu(h @ w2 + b2)
    gemm_wmma<128, 4, 1, false><<<(M / 16) * (DP / 64) / 8, 256, 0, stream>>>(
        h16, w2p, b2, nullptr, gbuf, DP);

    // causal depthwise conv + silu
    conv_silu_kernel<<<(M * DP) / 256, 256, 0, stream>>>(x1pre, conv_w, conv_b,
                                                         x1, x116, L);

    // B, C projections (N=32 -> 16x32 tiles) and delta = softplus(x1 @ wd + bd)
    gemm_wmma<256, 2, 0, false><<<(M / 16) * (NS / 32) / 8, 256, 0, stream>>>(
        x116, wBp, bB, nullptr, Bmat, NS);
    gemm_wmma<256, 2, 0, false><<<(M / 16) * (NS / 32) / 8, 256, 0, stream>>>(
        x116, wCp, bC, nullptr, Cmat, NS);
    gemm_wmma<256, 4, 2, false><<<(M / 16) * (DP / 64) / 8, 256, 0, stream>>>(
        x116, wdp, bd, nullptr, dbuf, DP);

    // chunked selective scan: local scans -> chunk carry -> final scan + gate
    const int nwaves = B * DP * SCAN_NC;  // 16384
    scan_pass1<<<nwaves / 8, 256, 0, stream>>>(x1, dbuf, Bmat, A_log, Sbuf, Pbuf, L);
    scan_combine<<<(B * DP * NS) / 256, 256, 0, stream>>>(Sbuf, Pbuf, Ibuf);
    scan_pass2<<<nwaves / 8, 256, 0, stream>>>(x1, dbuf, Bmat, Cmat, A_log, Dv,
                                               gbuf, Ibuf, prod16, L);

    // out = prod @ w3 + b3 + x (residual)
    gemm_wmma<256, 4, 0, true><<<(M / 16) * (DIN / 64) / 8, 256, 0, stream>>>(
        prod16, w3p, b3, x, (float*)d_out, DIN);
}
